// ConvAutoEncoder_83227876261959
// MI455X (gfx1250) — matrix operand order repaired and checked by reading, not compiled, not run on hardware
//
#include <hip/hip_runtime.h>
#include <hip/hip_bf16.h>
#include <math.h>

typedef __attribute__((ext_vector_type(16))) _Float16 v16h;
typedef __attribute__((ext_vector_type(8)))  _Float16 v8h;
typedef __attribute__((ext_vector_type(8)))  float    v8f;

#define BTOT 16384
#define NEG_SLOPE 0.2f

// ---------------------------------------------------------------------------
// Packed weight workspace layout (units: _Float16):
//   G1 (conv-as-GEMM, K=256,N=128, KS=8,NT=8):  vectors [nt][ks][lane][16h]  at vec 0
//   G2 (dec_w1,       K=128,N=512, KS=4,NT=32): at vec 2048
//   G3 (dec_w2,       K=512,N=128, KS=16,NT=8): at vec 6144
// One "vector" = 16 halfs = what one lane feeds a single WMMA B operand step.
// Lane l holds column n = nt*16 + (l&15), k = ks*32 + (l>=16 ? 8 : 0) + j + (j>=8 ? 8 : 0).
// ---------------------------------------------------------------------------
__global__ __launch_bounds__(256) void pack_weights_kernel(
    const float* __restrict__ conv_w,   // (64,64,3)
    const float* __restrict__ dec_w1,   // (512,128)
    const float* __restrict__ dec_w2,   // (128,512)
    _Float16* __restrict__ ws)
{
    int v = blockIdx.x * 256 + threadIdx.x;       // 0..10239
    if (v >= 10240) return;
    int mat, nt, ks, lane;
    if (v < 2048) {                 // G1: nt stride 8*32=256
        mat = 0; nt = v >> 8; ks = (v >> 5) & 7;  lane = v & 31;
    } else if (v < 6144) {          // G2: nt stride 4*32=128
        int r = v - 2048;
        mat = 1; nt = r >> 7; ks = (r >> 5) & 3;  lane = r & 31;
    } else {                        // G3: nt stride 16*32=512
        int r = v - 6144;
        mat = 2; nt = r >> 9; ks = (r >> 5) & 15; lane = r & 31;
    }
    int n     = nt * 16 + (lane & 15);
    int kbase = ks * 32 + ((lane >> 4) << 3);
    _Float16* dst = ws + (size_t)v * 16;
    #pragma unroll
    for (int j = 0; j < 16; ++j) {
        int k = kbase + j + ((j >> 3) << 3);
        float val;
        if (mat == 0) {
            // conv as GEMM: k = i*4+q (channel i, spatial q), n = o*2+p
            int i = k >> 2, q = k & 3, o = n >> 1, p = n & 1, r = q - p;
            val = (r >= 0 && r < 3) ? conv_w[(o * 64 + i) * 3 + r] : 0.0f;
        } else if (mat == 1) {
            val = dec_w1[n * 128 + k];     // B[k][n] = dec_w1[n,k]
        } else {
            val = dec_w2[n * 512 + k];     // B[k][n] = dec_w2[n,k]
        }
        dst[j] = (_Float16)val;
    }
}

// ---------------------------------------------------------------------------
// Fused kernel: 16 batches per block, 256 threads (8 waves).
// ---------------------------------------------------------------------------
__global__ __launch_bounds__(256) void fused_gat_conv_dec_kernel(
    const float* __restrict__ x,        // (B,64,2)
    const float* __restrict__ W_gat,    // (4,2)
    const float* __restrict__ a_attn,   // (1,8)
    const float* __restrict__ conv_b,   // (64)
    const float* __restrict__ dec_b1,   // (512)
    const float* __restrict__ dec_b2,   // (128)
    const _Float16* __restrict__ wsB,   // packed weights
    float* __restrict__ out)            // [B*128 encoder_out | B*128 decoded]
{
    __shared__ float     zbuf[16][64][4];   // 16 KB
    __shared__ float     esb[16][64];       //  4 KB
    __shared__ float     edb[16][64];       //  4 KB
    __shared__ _Float16  A1[16][264];       // h,  K=256  (+8 pad: stride%64dw==4)
    __shared__ _Float16  A2[16][136];       // flat, K=128 (+8 pad)
    __shared__ _Float16  A3[16][520];       // hdec, K=512 (+8 pad)
    __shared__ float     bias[704];         // [0,64)=conv_b [64,576)=dec_b1 [576,704)=dec_b2

    const int tid   = threadIdx.x;
    const int bBase = blockIdx.x * 16;

    for (int i = tid; i < 704; i += 256)
        bias[i] = (i < 64) ? conv_b[i] : ((i < 576) ? dec_b1[i - 64] : dec_b2[i - 576]);

    float Wg[8], aa[8];
    #pragma unroll
    for (int i = 0; i < 8; ++i) { Wg[i] = W_gat[i]; aa[i] = a_attn[i]; }

    // ---- Phase 1a: z, es, ed --------------------------------------------
    #pragma unroll
    for (int it = 0; it < 4; ++it) {
        int idx = it * 256 + tid;
        int bl = idx >> 6, t = idx & 63;
        size_t xo = (size_t)(bBase + bl) * 128 + t * 2;
        float x0 = x[xo], x1 = x[xo + 1];
        float es = 0.f, ed = 0.f;
        #pragma unroll
        for (int d = 0; d < 4; ++d) {
            float zd = x0 * Wg[d * 2] + x1 * Wg[d * 2 + 1];
            zbuf[bl][t][d] = zd;
            es += zd * aa[d];
            ed += zd * aa[4 + d];
        }
        esb[bl][t] = es;
        edb[bl][t] = ed;
    }
    __syncthreads();

    // ---- Phase 1b: masked leaky-relu softmax over i, h -> A1 (f16) ------
    #pragma unroll 1
    for (int it = 0; it < 4; ++it) {
        int idx = it * 256 + tid;
        int bl = idx >> 6, j = idx & 63;
        float edj = edb[bl][j];
        float m = -3.0e38f;
        for (int i = 0; i < 64; ++i) {
            if (i == j) continue;
            float e = esb[bl][i] + edj;
            e = (e > 0.f) ? e : NEG_SLOPE * e;
            m = fmaxf(m, e);
        }
        float den = 0.f, n0 = 0.f, n1 = 0.f, n2 = 0.f, n3 = 0.f;
        for (int i = 0; i < 64; ++i) {
            if (i == j) continue;
            float e = esb[bl][i] + edj;
            e = (e > 0.f) ? e : NEG_SLOPE * e;
            float w = __expf(e - m);
            den += w;
            n0 += w * zbuf[bl][i][0];
            n1 += w * zbuf[bl][i][1];
            n2 += w * zbuf[bl][i][2];
            n3 += w * zbuf[bl][i][3];
        }
        float inv = 1.0f / den;
        A1[bl][j * 4 + 0] = (_Float16)(n0 * inv);
        A1[bl][j * 4 + 1] = (_Float16)(n1 * inv);
        A1[bl][j * 4 + 2] = (_Float16)(n2 * inv);
        A1[bl][j * 4 + 3] = (_Float16)(n3 * inv);
    }
    __syncthreads();

    // ---- WMMA GEMM phases -----------------------------------------------
    const int wave  = tid >> 5;           // 0..7
    const int lane  = tid & 31;
    const int mrow  = lane & 15;          // M index within first half
    const int khalf = (lane >> 4) << 3;   // lanes 16..31 take K offset +8
    const int mhi   = (lane >> 4) << 3;   // C frag: M = r + (lane>=16 ? 8 : 0)

    // A fragment loader: 16-bit A layout = elems 0..7 -> k0..k0+7, 8..15 -> k0+16..k0+23
    auto loadA = [&](const _Float16* buf, int stride, int ks) -> v16h {
        const _Float16* p = buf + mrow * stride + ks * 32 + khalf;
        v8h lo = *(const v8h*)(p);
        v8h hi = *(const v8h*)(p + 16);
        v16h a;
        #pragma unroll
        for (int j = 0; j < 8; ++j) { a[j] = lo[j]; a[8 + j] = hi[j]; }
        return a;
    };
    auto loadB = [&](int vecbase, int nt, int KS, int ks) -> v16h {
        return *(const v16h*)(wsB + ((size_t)(vecbase + (nt * KS + ks) * 32 + lane)) * 16);
    };

    // GEMM1: conv-as-GEMM  (K=256 -> N=128), one N-tile per wave
    {
        int nt = wave;
        v8f c = {};
        #pragma unroll
        for (int ks = 0; ks < 8; ++ks) {
            v16h a = loadA(&A1[0][0], 264, ks);
            v16h b = loadB(0, nt, 8, ks);
            c = __builtin_amdgcn_wmma_f32_16x16x32_f16(false, a, false, b, (short)0, c, false, false);
        }
        int n = nt * 16 + (lane & 15);
        float cb = bias[n >> 1];           // conv_b[o], n = 2o+p
        #pragma unroll
        for (int r = 0; r < 8; ++r) {
            int m = r + mhi;
            float v = c[r] + cb;
            out[(size_t)(bBase + m) * 128 + n] = v;     // encoder_out (f32)
            A2[m][n] = (_Float16)v;                     // decoder input
        }
    }
    __syncthreads();

    // GEMM2: dec layer 1 (K=128 -> N=512) + relu, 4 N-tiles per wave
    #pragma unroll 1
    for (int q = 0; q < 4; ++q) {
        int nt = wave * 4 + q;
        v8f c = {};
        #pragma unroll
        for (int ks = 0; ks < 4; ++ks) {
            v16h a = loadA(&A2[0][0], 136, ks);
            v16h b = loadB(2048, nt, 4, ks);
            c = __builtin_amdgcn_wmma_f32_16x16x32_f16(false, a, false, b, (short)0, c, false, false);
        }
        int n = nt * 16 + (lane & 15);
        float bb = bias[64 + n];
        #pragma unroll
        for (int r = 0; r < 8; ++r) {
            int m = r + mhi;
            float v = c[r] + bb;
            v = (v > 0.f) ? v : 0.f;
            A3[m][n] = (_Float16)v;
        }
    }
    __syncthreads();

    // GEMM3: dec layer 2 (K=512 -> N=128), one N-tile per wave
    {
        int nt = wave;
        v8f c = {};
        #pragma unroll
        for (int ks = 0; ks < 16; ++ks) {
            v16h a = loadA(&A3[0][0], 520, ks);
            v16h b = loadB(6144, nt, 16, ks);
            c = __builtin_amdgcn_wmma_f32_16x16x32_f16(false, a, false, b, (short)0, c, false, false);
        }
        int n = nt * 16 + (lane & 15);
        float bb = bias[576 + n];
        #pragma unroll
        for (int r = 0; r < 8; ++r) {
            int m = r + mhi;
            out[(size_t)BTOT * 128 + (size_t)(bBase + m) * 128 + n] = c[r] + bb;
        }
    }
}

extern "C" void kernel_launch(void* const* d_in, const int* in_sizes, int n_in,
                              void* d_out, int out_size, void* d_ws, size_t ws_size,
                              hipStream_t stream) {
    const float* x      = (const float*)d_in[0];
    const float* W_gat  = (const float*)d_in[1];
    const float* a_attn = (const float*)d_in[2];
    const float* conv_w = (const float*)d_in[3];
    const float* conv_b = (const float*)d_in[4];
    const float* dec_w1 = (const float*)d_in[5];
    const float* dec_b1 = (const float*)d_in[6];
    const float* dec_w2 = (const float*)d_in[7];
    const float* dec_b2 = (const float*)d_in[8];
    _Float16* ws = (_Float16*)d_ws;

    pack_weights_kernel<<<40, 256, 0, stream>>>(conv_w, dec_w1, dec_w2, ws);
    fused_gat_conv_dec_kernel<<<BTOT / 16, 256, 0, stream>>>(
        x, W_gat, a_attn, conv_b, dec_b1, dec_b2, ws, (float*)d_out);
}
